// Similarity_42202348650844
// MI455X (gfx1250) — matrix operand order repaired and checked by reading, not compiled, not run on hardware
//
#include <hip/hip_runtime.h>
#include <hip/hip_bf16.h>

// Problem dimensions (fixed by the reference)
#define B_DIM 64
#define C_DIM 2048
#define Q_DIM 512
#define D_DIM 512

// Tiling
#define BM 128         // C-rows per block
#define BN 128         // Q-cols per block
#define TILE_K 32      // bf16 WMMA K
#define LDS_STRIDE 40  // bf16 elems per LDS tile row (32 + 8 pad -> 80B, 16B aligned)

typedef __attribute__((ext_vector_type(16))) __bf16 v16bf;
typedef __attribute__((ext_vector_type(8)))  __bf16 v8bf;
typedef __attribute__((ext_vector_type(8)))  float  v8f;

union Frag16 { v16bf v; v8bf h[2]; };

// ---------------------------------------------------------------------------
// rowdot: out[row] = dot(x[row, 0:d], w[0:d]);   one wave32 per row
// ---------------------------------------------------------------------------
__global__ __launch_bounds__(256) void rowdot_kernel(
    const float* __restrict__ x, const float* __restrict__ w,
    float* __restrict__ outv, int nrows, int d)
{
    const int row  = blockIdx.x * 8 + (threadIdx.x >> 5);
    const int lane = threadIdx.x & 31;
    if (row >= nrows) return;
    const float* p = x + (size_t)row * d;
    float s = 0.f;
    for (int i = lane * 4; i < d; i += 128) {
        float4 v  = *(const float4*)(p + i);
        float4 ww = *(const float4*)(w + i);
        s += v.x * ww.x + v.y * ww.y + v.z * ww.z + v.w * ww.w;
    }
#pragma unroll
    for (int off = 16; off > 0; off >>= 1) s += __shfl_down(s, off, 32);
    if (lane == 0) outv[row] = s;
}

// ---------------------------------------------------------------------------
// Main: sim[b,c,q] = (ctx[b,c,:] . (qry[b,q,:]*wqc)) + xq[b,q] + xc[b,c],
//       then mask multiply and +inf -> -inf, exactly like the reference.
// ---------------------------------------------------------------------------
__global__ __launch_bounds__(256) void sim_wmma_kernel(
    const float* __restrict__ context,   // [B,C,D]
    const float* __restrict__ query,     // [B,Q,D]
    const float* __restrict__ wqc,       // [D]
    const int*   __restrict__ cmask,     // [B,C]
    const int*   __restrict__ qmask,     // [B,Q]
    const float* __restrict__ xc,        // [B,C]
    const float* __restrict__ xq,        // [B,Q]
    float* __restrict__ out)             // [B,C,Q]
{
    __shared__ __bf16 Atile[BM * LDS_STRIDE];  // context tile (rows = c)
    __shared__ __bf16 Btile[BN * LDS_STRIDE];  // (query*wqc) tile (rows = q)

    const int b  = blockIdx.z;
    const int c0 = blockIdx.y * BM;
    const int q0 = blockIdx.x * BN;

    const int tid   = threadIdx.x;
    const int lane  = tid & 31;
    const int wave  = tid >> 5;
    const int waveM = (wave & 1) * 64;   // 2 waves tile M (64 rows each)
    const int waveN = (wave >> 1) * 32;  // 4 waves tile N (32 cols each)

    const int lr = lane & 15;            // fragment row (A) / col (B) / N (D)
    const int lh = lane >> 4;            // K-half / M-half selector

    const float* ctxB = context + (size_t)b * C_DIM * D_DIM;
    const float* qryB = query   + (size_t)b * Q_DIM * D_DIM;

    v8f acc[4][2];
#pragma unroll
    for (int mi = 0; mi < 4; ++mi)
#pragma unroll
        for (int ni = 0; ni < 2; ++ni)
#pragma unroll
            for (int e = 0; e < 8; ++e) acc[mi][ni][e] = 0.f;

    const int loadRow  = tid >> 1;        // 0..127
    const int loadCol0 = (tid & 1) * 16;  // 0 or 16

    for (int k0 = 0; k0 < D_DIM; k0 += TILE_K) {
        // ---- stage A (context) tile: f32 -> bf16 ----
        {
            const float* src = ctxB + (size_t)(c0 + loadRow) * D_DIM + k0 + loadCol0;
            float4 f0 = ((const float4*)src)[0];
            float4 f1 = ((const float4*)src)[1];
            float4 f2 = ((const float4*)src)[2];
            float4 f3 = ((const float4*)src)[3];
            __bf16* dst = &Atile[loadRow * LDS_STRIDE + loadCol0];
            v8bf p0 = {(__bf16)f0.x, (__bf16)f0.y, (__bf16)f0.z, (__bf16)f0.w,
                       (__bf16)f1.x, (__bf16)f1.y, (__bf16)f1.z, (__bf16)f1.w};
            v8bf p1 = {(__bf16)f2.x, (__bf16)f2.y, (__bf16)f2.z, (__bf16)f2.w,
                       (__bf16)f3.x, (__bf16)f3.y, (__bf16)f3.z, (__bf16)f3.w};
            *(v8bf*)(dst)     = p0;
            *(v8bf*)(dst + 8) = p1;
        }
        // ---- stage B (query * wqc) tile: f32 -> bf16 ----
        {
            const float* src = qryB + (size_t)(q0 + loadRow) * D_DIM + k0 + loadCol0;
            const float* ws  = wqc + k0 + loadCol0;
            float4 f0 = ((const float4*)src)[0];
            float4 f1 = ((const float4*)src)[1];
            float4 f2 = ((const float4*)src)[2];
            float4 f3 = ((const float4*)src)[3];
            float4 w0 = ((const float4*)ws)[0];
            float4 w1 = ((const float4*)ws)[1];
            float4 w2 = ((const float4*)ws)[2];
            float4 w3 = ((const float4*)ws)[3];
            __bf16* dst = &Btile[loadRow * LDS_STRIDE + loadCol0];
            v8bf p0 = {(__bf16)(f0.x * w0.x), (__bf16)(f0.y * w0.y),
                       (__bf16)(f0.z * w0.z), (__bf16)(f0.w * w0.w),
                       (__bf16)(f1.x * w1.x), (__bf16)(f1.y * w1.y),
                       (__bf16)(f1.z * w1.z), (__bf16)(f1.w * w1.w)};
            v8bf p1 = {(__bf16)(f2.x * w2.x), (__bf16)(f2.y * w2.y),
                       (__bf16)(f2.z * w2.z), (__bf16)(f2.w * w2.w),
                       (__bf16)(f3.x * w3.x), (__bf16)(f3.y * w3.y),
                       (__bf16)(f3.z * w3.z), (__bf16)(f3.w * w3.w)};
            *(v8bf*)(dst)     = p0;
            *(v8bf*)(dst + 8) = p1;
        }
        __syncthreads();

        // ---- load fragments (CDNA5 16-bit A 16x32 / B 32x16 layouts) ----
        v16bf afrag[4];
#pragma unroll
        for (int mi = 0; mi < 4; ++mi) {
            const __bf16* ap = &Atile[(waveM + mi * 16 + lr) * LDS_STRIDE + 8 * lh];
            Frag16 u;
            u.h[0] = *(const v8bf*)(ap);       // K = 8h .. 8h+7
            u.h[1] = *(const v8bf*)(ap + 16);  // K = 16+8h .. 16+8h+7
            afrag[mi] = u.v;
        }
        v16bf bfrag[2];
#pragma unroll
        for (int ni = 0; ni < 2; ++ni) {
            const __bf16* bp = &Btile[(waveN + ni * 16 + lr) * LDS_STRIDE + 8 * lh];
            Frag16 u;
            u.h[0] = *(const v8bf*)(bp);       // K = 8h .. 8h+7
            u.h[1] = *(const v8bf*)(bp + 16);  // K = 16+8h .. 16+8h+7
            bfrag[ni] = u.v;
        }

        // ---- 8 WMMAs per wave per K-step ----
#pragma unroll
        for (int mi = 0; mi < 4; ++mi)
#pragma unroll
            for (int ni = 0; ni < 2; ++ni)
                acc[mi][ni] = __builtin_amdgcn_wmma_f32_16x16x32_bf16(
                    /*neg_a=*/false, afrag[mi], /*neg_b=*/false, bfrag[ni],
                    /*c_mod=*/(short)0, acc[mi][ni],
                    /*reuse_a=*/false, /*reuse_b=*/false);

        __syncthreads();
    }

    // ---- epilogue: + xq + xc, mask multiply, +inf -> -inf ----
    const float NEG_INF = -__builtin_inff();
    const float POS_INF =  __builtin_inff();
    const size_t bC = (size_t)b * C_DIM;
    const size_t bQ = (size_t)b * Q_DIM;

#pragma unroll
    for (int ni = 0; ni < 2; ++ni) {
        const int q = q0 + waveN + ni * 16 + lr;   // N = lane & 15
        const float xqv = xq[bQ + q];
        const float qm  = (qmask[bQ + q] == 1) ? 1.f : NEG_INF;
#pragma unroll
        for (int mi = 0; mi < 4; ++mi) {
            const int cbase = c0 + waveM + mi * 16 + 8 * lh;  // M = v + 8h
            float4 xcv0 = *(const float4*)&xc[bC + cbase];
            float4 xcv1 = *(const float4*)&xc[bC + cbase + 4];
            int4   cmk0 = *(const int4*)&cmask[bC + cbase];
            int4   cmk1 = *(const int4*)&cmask[bC + cbase + 4];
            const float xcs[8] = {xcv0.x, xcv0.y, xcv0.z, xcv0.w,
                                  xcv1.x, xcv1.y, xcv1.z, xcv1.w};
            const int   cms[8] = {cmk0.x, cmk0.y, cmk0.z, cmk0.w,
                                  cmk1.x, cmk1.y, cmk1.z, cmk1.w};
#pragma unroll
            for (int v = 0; v < 8; ++v) {
                const int c = cbase + v;
                float s = acc[mi][ni][v] + xqv + xcs[v];
                const float cm = (cms[v] == 1) ? 1.f : NEG_INF;
                s = s * cm * qm;
                if (s == POS_INF) s = NEG_INF;
                out[((size_t)bC + c) * Q_DIM + q] = s;
            }
        }
    }
}

// ---------------------------------------------------------------------------
extern "C" void kernel_launch(void* const* d_in, const int* in_sizes, int n_in,
                              void* d_out, int out_size, void* d_ws, size_t ws_size,
                              hipStream_t stream) {
    (void)in_sizes; (void)n_in; (void)out_size; (void)ws_size;
    const float* context = (const float*)d_in[0];  // [B,C,D]
    const int*   cmask   = (const int*)  d_in[1];  // [B,C]
    const float* query   = (const float*)d_in[2];  // [B,Q,D]
    const int*   qmask   = (const int*)  d_in[3];  // [B,Q]
    const float* wq      = (const float*)d_in[4];  // [D]
    const float* wc      = (const float*)d_in[5];  // [D]
    const float* wqc     = (const float*)d_in[6];  // [D]
    float* out = (float*)d_out;

    float* xq = (float*)d_ws;                      // [B,Q]
    float* xc = xq + (size_t)B_DIM * Q_DIM;        // [B,C]

    // x_q[b,q] = query[b,q,:] . wq
    {
        const int nrows = B_DIM * Q_DIM;
        rowdot_kernel<<<nrows / 8, 256, 0, stream>>>(query, wq, xq, nrows, D_DIM);
    }
    // x_c[b,c] = context[b,c,:] . wc
    {
        const int nrows = B_DIM * C_DIM;
        rowdot_kernel<<<nrows / 8, 256, 0, stream>>>(context, wc, xc, nrows, D_DIM);
    }
    // main fused GEMM + epilogue
    dim3 grid(Q_DIM / BN, C_DIM / BM, B_DIM);
    sim_wmma_kernel<<<grid, 256, 0, stream>>>(context, query, wqc,
                                              cmask, qmask, xc, xq, out);
}
